// Messages_16003048145211
// MI455X (gfx1250) — matrix operand order repaired and checked by reading, not compiled, not run on hardware
//
#include <hip/hip_runtime.h>
#include <hip/hip_fp16.h>

typedef __attribute__((ext_vector_type(16))) _Float16 v16h;
typedef __attribute__((ext_vector_type(8)))  _Float16 v8h;
typedef __attribute__((ext_vector_type(8)))  float    v8f;

#define WAVES 4

// ---------------------------------------------------------------------------
// Wave-local sync: LDS staging buffers are per-wave; CDNA5 keeps same-wave LDS
// ops in order, so we only need to stop the compiler from reordering across
// the producer/consumer boundary (no s_barrier needed).
// ---------------------------------------------------------------------------
static __device__ __forceinline__ void wave_sync() {
  __builtin_amdgcn_fence(__ATOMIC_RELEASE, "wavefront");
  __builtin_amdgcn_wave_barrier();
  __builtin_amdgcn_fence(__ATOMIC_ACQUIRE, "wavefront");
}

// ---------------------------------------------------------------------------
// WMMA helpers (CDNA5 16x16x32 f16 -> f32)
// ---------------------------------------------------------------------------
static __device__ __forceinline__ v8f wmma_f16(v16h a, v16h b, v8f c) {
  // (neg_a, A, neg_b, B, c_mod, C, reuse_a, reuse_b)
  return __builtin_amdgcn_wmma_f32_16x16x32_f16(false, a, false, b, (short)0, c,
                                                false, false);
}

// Load an A operand (16 rows x 32 K) from a row-major f16 LDS buffer.
// k order used here (contiguous, matches B packing): lanes 0-15 hold row M=lane, K=(0..15);
//                    lanes 16-31 hold row M=lane-16, K=(16..31).
static __device__ __forceinline__ v16h load_A_lds(const _Float16* buf, int stride,
                                                  int kb) {
  const int lane = threadIdx.x & 31;
  const int m = lane & 15;
  const int kbase = kb * 32 + ((lane & 16) ? 16 : 0);
  const v8h lo = *(const v8h*)(buf + m * stride + kbase);
  const v8h hi = *(const v8h*)(buf + m * stride + kbase + 8);
  v16h a;
#pragma unroll
  for (int i = 0; i < 8; ++i) { a[i] = lo[i]; a[8 + i] = hi[i]; }
  return a;
}

// Store a C accumulator tile (16x16 f32) into a row-major f16 LDS buffer.
// C layout: lane holds col N=lane%16, rows g + (lane<16 ? 0 : 8).
template <bool ACT>
static __device__ __forceinline__ void store_C_lds(_Float16* buf, int stride,
                                                   int nb, v8f acc, float bias) {
  const int lane = threadIdx.x & 31;
  const int col = nb * 16 + (lane & 15);
  const int r0 = (lane & 16) ? 8 : 0;
#pragma unroll
  for (int g = 0; g < 8; ++g) {
    float v = acc[g];
    if (ACT) {
      v += bias;
      v = (v >= 0.0f) ? v : 0.1f * v;
    }
    buf[(r0 + g) * stride + col] = (_Float16)v;
  }
}

// Pack 8 floats into one 16-byte LDS store (lane's channels are contiguous).
static __device__ __forceinline__ void store8h(_Float16* dst, const float* v) {
  v8h p;
#pragma unroll
  for (int i = 0; i < 8; ++i) p[i] = (_Float16)v[i];
  *(v8h*)dst = p;
}

// ---------------------------------------------------------------------------
// Zero the output buffer (graph-capture safe).
// ---------------------------------------------------------------------------
__global__ void zero_kernel(float* __restrict__ p, int n) {
  for (int i = blockIdx.x * blockDim.x + threadIdx.x; i < n;
       i += gridDim.x * blockDim.x)
    p[i] = 0.0f;
}

// ---------------------------------------------------------------------------
// Per-node channel projections: P = _tl(W_L*, x)   (16 output channels)
// ---------------------------------------------------------------------------
__global__ void node_proj_kernel(const float* __restrict__ x_a,
                                 const float* __restrict__ x_v,
                                 const float* __restrict__ x_d,
                                 const float* __restrict__ W_L0,
                                 const float* __restrict__ W_L1,
                                 const float* __restrict__ W_L2,
                                 float* __restrict__ Pa, float* __restrict__ Pv,
                                 float* __restrict__ Pd, int N) {
  int t = blockIdx.x * blockDim.x + threadIdx.x;
  if (t >= N * 16) return;
  const int n = t >> 4, c = t & 15;
  float acc = 0.0f;
  for (int a = 0; a < 64; ++a) acc += W_L0[c * 64 + a] * x_a[n * 64 + a];
  Pa[t] = acc;
  float av0 = 0.f, av1 = 0.f, av2 = 0.f;
  for (int a = 0; a < 32; ++a) {
    const float w = W_L1[c * 32 + a];
    av0 += w * x_v[(n * 32 + a) * 3 + 0];
    av1 += w * x_v[(n * 32 + a) * 3 + 1];
    av2 += w * x_v[(n * 32 + a) * 3 + 2];
  }
  Pv[t * 3 + 0] = av0; Pv[t * 3 + 1] = av1; Pv[t * 3 + 2] = av2;
  float ad[9];
#pragma unroll
  for (int k = 0; k < 9; ++k) ad[k] = 0.0f;
  for (int a = 0; a < 16; ++a) {
    const float w = W_L2[c * 16 + a];
#pragma unroll
    for (int k = 0; k < 9; ++k) ad[k] += w * x_d[(n * 16 + a) * 9 + k];
  }
#pragma unroll
  for (int k = 0; k < 9; ++k) Pd[t * 9 + k] = ad[k];
}

// ---------------------------------------------------------------------------
// Pack W^T into WMMA B-operand lane order, f16, zero K/N padding.
// B layout per 32x16 K-block: lane holds col N=lane%16; element i holds
// K = kb*32 + (lane<16 ? 0 : 16) + i.  Storage: [kb][nb][lane][elem].
// ncat==1: w0 is (Nreal, Kreal) row-major (B[k][n] = w0[n*Kreal+k]).
// ncat>1 : k selects among ncat weights, each (Nreal, 16) row-major.
// ---------------------------------------------------------------------------
__global__ void pack_weights(const float* w0, const float* w1, const float* w2,
                             const float* w3, const float* w4, int ncat, int Kreal,
                             int Nreal, int KB, int NB, _Float16* __restrict__ out) {
  const int tot = KB * NB * 512;
  for (int o = blockIdx.x * blockDim.x + threadIdx.x; o < tot;
       o += gridDim.x * blockDim.x) {
    const int elem = o & 15;
    const int lane = (o >> 4) & 31;
    const int blk = o >> 9;
    const int nb = blk % NB;
    const int kb = blk / NB;
    const int k = kb * 32 + ((lane & 16) ? 16 : 0) + elem;
    const int n = nb * 16 + (lane & 15);
    float val = 0.0f;
    if (k < Kreal && n < Nreal) {
      if (ncat == 1) {
        val = w0[n * Kreal + k];
      } else {
        const float* w = (k < 16) ? w0 : (k < 32) ? w1 : (k < 48) ? w2
                                       : (k < 64) ? w3 : w4;
        val = w[n * 16 + (k & 15)];
      }
    }
    out[o] = (_Float16)val;
  }
}

// ---------------------------------------------------------------------------
// Fused per-edge kernel: 4 waves/block, one 16-edge tile per wave.
// ---------------------------------------------------------------------------
__global__ __launch_bounds__(128) void edge_kernel(
    const float* __restrict__ r_ij, const int* __restrict__ dst,
    const int* __restrict__ src, const float* __restrict__ W_enc,
    const float* __restrict__ b_enc, const float* __restrict__ b1,
    const float* __restrict__ b2, const float* __restrict__ b3,
    const float* __restrict__ Pa, const float* __restrict__ Pv,
    const float* __restrict__ Pd, const _Float16* __restrict__ WBa,
    const _Float16* __restrict__ WB1, const _Float16* __restrict__ WB2,
    const _Float16* __restrict__ WBdir, const _Float16* __restrict__ WB3,
    const _Float16* __restrict__ WBv, const _Float16* __restrict__ WBd,
    float* __restrict__ outA, float* __restrict__ outV, float* __restrict__ outD,
    int E) {
  __shared__ __align__(32) _Float16 sY[WAVES][16 * 96];    // Y staging (K<=96)
  __shared__ __align__(32) _Float16 sPsi[WAVES][16 * 64];  // psi_a pre (f16)
  __shared__ __align__(32) _Float16 sH1[WAVES][16 * 128];  // MLP hidden 1
  __shared__ __align__(32) _Float16 sH2[WAVES][16 * 128];  // MLP hidden 2

  const int wave = threadIdx.x >> 5;
  const int lane = threadIdx.x & 31;
  const int tile = blockIdx.x * WAVES + wave;
  const int m = lane & 15;              // edge row within tile
  const int ch0 = (lane & 16) ? 8 : 0;  // this lane's channel half
  const int e = tile * 16 + m;
  const int ec = (e < E) ? e : (E - 1);

  _Float16* Yb = &sY[wave][0];
  _Float16* Pb = &sPsi[wave][0];
  _Float16* H1 = &sH1[wave][0];
  _Float16* H2 = &sH2[wave][0];
  const v16h* vWBa = (const v16h*)WBa;
  const v16h* vWB1 = (const v16h*)WB1;
  const v16h* vWB2 = (const v16h*)WB2;
  const v16h* vWBdir = (const v16h*)WBdir;
  const v16h* vWB3 = (const v16h*)WB3;
  const v16h* vWBv = (const v16h*)WBv;
  const v16h* vWBd = (const v16h*)WBd;

  // Zero the K-pad region [48,96) once (data regions never reach k>=80).
  {
    v8h z;
#pragma unroll
    for (int i = 0; i < 8; ++i) z[i] = (_Float16)0.0f;
    for (int t = lane; t < 16 * 6; t += 32) {
      const int rw = t / 6, seg = t % 6;
      *(v8h*)(Yb + rw * 96 + 48 + seg * 8) = z;
    }
  }

  // Prefetch upcoming edge data into cache hierarchy (global_prefetch_b8).
  __builtin_prefetch(&r_ij[(size_t)(ec + 16) * 3], 0, 1);
  __builtin_prefetch(&dst[(ec + 16)], 0, 1);

  // ---------------- per-edge scalar stage (radial + gathers) ----------------
  const float rx = r_ij[ec * 3 + 0], ry = r_ij[ec * 3 + 1], rz = r_ij[ec * 3 + 2];
  const float d = sqrtf(rx * rx + ry * ry + rz * rz);
  float rbf[8];
  const float x8 = d * 8.0f;
#pragma unroll
  for (int mm = 0; mm < 8; ++mm) {
    const float t = x8 - (float)mm;
    rbf[mm] = __expf(-0.5f * t * t);
  }
  const float n7 = 7.0f * d;
  const float sg = tanhf(n7) / (n7 + 1e-12f);
  const float rr[3] = {7.0f * rx * sg, 7.0f * ry * sg, 7.0f * rz * sg};
  const int jn = dst[ec];
  const float* pap = Pa + (size_t)jn * 16;
  const float* pvp = Pv + (size_t)jn * 48;
  const float* pdp = Pd + (size_t)jn * 144;

  float rad[8], pa[8], pv[8][3], pvr[8];
  {
    float y0[8], y1[8], y2[8];
#pragma unroll
    for (int c8 = 0; c8 < 8; ++c8) {
      const int c = ch0 + c8;
      float rd = b_enc[c];
#pragma unroll
      for (int mm = 0; mm < 8; ++mm) rd += rbf[mm] * W_enc[c * 8 + mm];
      rad[c8] = rd;
      pa[c8] = pap[c];
      pv[c8][0] = pvp[c * 3 + 0];
      pv[c8][1] = pvp[c * 3 + 1];
      pv[c8][2] = pvp[c * 3 + 2];
      pvr[c8] = pv[c8][0] * rr[0] + pv[c8][1] * rr[1] + pv[c8][2] * rr[2];
      float a220 = 0.0f;
#pragma unroll
      for (int i = 0; i < 3; ++i) {
        const float pri = pdp[c * 9 + i * 3 + 0] * rr[0] +
                          pdp[c * 9 + i * 3 + 1] * rr[1] +
                          pdp[c * 9 + i * 3 + 2] * rr[2];
        a220 += rr[i] * pri;
      }
      y0[c8] = pa[c8] * rd;   // y000
      y1[c8] = pvr[c8] * rd;  // y110
      y2[c8] = a220 * rd;     // y220
    }
    // Y_a = [y000 | y110 | y220]  (K = 48, padded to 64)
    store8h(Yb + m * 96 + 0 + ch0, y0);
    store8h(Yb + m * 96 + 16 + ch0, y1);
    store8h(Yb + m * 96 + 32 + ch0, y2);
  }
  wave_sync();

  v8f zero8;
#pragma unroll
  for (int g = 0; g < 8; ++g) zero8[g] = 0.0f;

  // ---- GEMM 1: psi_a_pre = Y_a @ Wcat^T   (K=64 pad, N=64) ----
  v8f accP[4];
#pragma unroll
  for (int nb = 0; nb < 4; ++nb) accP[nb] = zero8;
#pragma unroll
  for (int kb = 0; kb < 2; ++kb) {
    const v16h a = load_A_lds(Yb, 96, kb);
#pragma unroll
    for (int nb = 0; nb < 4; ++nb)
      accP[nb] = wmma_f16(a, vWBa[(kb * 4 + nb) * 32 + lane], accP[nb]);
  }
#pragma unroll
  for (int nb = 0; nb < 4; ++nb) store_C_lds<false>(Pb, 64, nb, accP[nb], 0.0f);
  wave_sync();

  // ---- GEMM 2: h1 = lrelu(psi_pre @ W1^T + b1)   (K=64, N=128) ----
  v8f acc8[8];
#pragma unroll
  for (int nb = 0; nb < 8; ++nb) acc8[nb] = zero8;
#pragma unroll
  for (int kb = 0; kb < 2; ++kb) {
    const v16h a = load_A_lds(Pb, 64, kb);
#pragma unroll
    for (int nb = 0; nb < 8; ++nb)
      acc8[nb] = wmma_f16(a, vWB1[(kb * 8 + nb) * 32 + lane], acc8[nb]);
  }
#pragma unroll
  for (int nb = 0; nb < 8; ++nb)
    store_C_lds<true>(H1, 128, nb, acc8[nb], b1[nb * 16 + (lane & 15)]);
  wave_sync();

  // ---- GEMM 3: h2 = lrelu(h1 @ W2^T + b2)   (K=128, N=128) ----
#pragma unroll
  for (int nb = 0; nb < 8; ++nb) acc8[nb] = zero8;
#pragma unroll
  for (int kb = 0; kb < 4; ++kb) {
    const v16h a = load_A_lds(H1, 128, kb);
#pragma unroll
    for (int nb = 0; nb < 8; ++nb)
      acc8[nb] = wmma_f16(a, vWB2[(kb * 8 + nb) * 32 + lane], acc8[nb]);
  }
#pragma unroll
  for (int nb = 0; nb < 8; ++nb)
    store_C_lds<true>(H2, 128, nb, acc8[nb], b2[nb * 16 + (lane & 15)]);
  wave_sync();

  // ---- Final: psi_a = psi_pre + b3 + psi_pre@Wdir^T + h2@W3^T ----
  v8f accF[4];
#pragma unroll
  for (int nb = 0; nb < 4; ++nb) {
    const float bv = b3[nb * 16 + (lane & 15)];
#pragma unroll
    for (int g = 0; g < 8; ++g) accF[nb][g] = accP[nb][g] + bv;
  }
#pragma unroll
  for (int kb = 0; kb < 2; ++kb) {
    const v16h a = load_A_lds(Pb, 64, kb);
#pragma unroll
    for (int nb = 0; nb < 4; ++nb)
      accF[nb] = wmma_f16(a, vWBdir[(kb * 4 + nb) * 32 + lane], accF[nb]);
  }
#pragma unroll
  for (int kb = 0; kb < 4; ++kb) {
    const v16h a = load_A_lds(H2, 128, kb);
#pragma unroll
    for (int nb = 0; nb < 4; ++nb)
      accF[nb] = wmma_f16(a, vWB3[(kb * 4 + nb) * 32 + lane], accF[nb]);
  }
  // scatter psi_a -> B_a[src]
  const int r0 = (lane & 16) ? 8 : 0;
#pragma unroll
  for (int nb = 0; nb < 4; ++nb) {
#pragma unroll
    for (int g = 0; g < 8; ++g) {
      const int ee = tile * 16 + r0 + g;
      if (ee < E) {
        const int sn = src[ee];
        unsafeAtomicAdd(&outA[(size_t)sn * 64 + nb * 16 + (lane & 15)],
                        0.1f * accF[nb][g]);
      }
    }
  }

  // ---- psi_v: 3 components, K=80 (pad 96), N=32 ----
#pragma unroll
  for (int i = 0; i < 3; ++i) {
    const int i1 = (i + 1) % 3, i2 = (i + 2) % 3;
    wave_sync();  // previous reads of Yb complete
    {
      float y0[8], y1[8], y2[8], y3[8], y4[8];
#pragma unroll
      for (int c8 = 0; c8 < 8; ++c8) {
        const int c = ch0 + c8;
        const float rd = rad[c8];
        const float pdr_i = pdp[c * 9 + i * 3 + 0] * rr[0] +
                            pdp[c * 9 + i * 3 + 1] * rr[1] +
                            pdp[c * 9 + i * 3 + 2] * rr[2];
        y0[c8] = pa[c8] * rd * rr[i];                                // y011
        y1[c8] = pv[c8][i] * rd;                                     // y101
        y2[c8] = rd * pvr[c8] * rr[i];                               // y121
        y3[c8] = rd * pdr_i;                                         // y211
        y4[c8] = rd * (pv[c8][i1] * rr[i2] - pv[c8][i2] * rr[i1]);   // y111
      }
      store8h(Yb + m * 96 + 0 + ch0, y0);
      store8h(Yb + m * 96 + 16 + ch0, y1);
      store8h(Yb + m * 96 + 32 + ch0, y2);
      store8h(Yb + m * 96 + 48 + ch0, y3);
      store8h(Yb + m * 96 + 64 + ch0, y4);
    }
    wave_sync();
    v8f aV[2];
#pragma unroll
    for (int nb = 0; nb < 2; ++nb) aV[nb] = zero8;
#pragma unroll
    for (int kb = 0; kb < 3; ++kb) {
      const v16h a = load_A_lds(Yb, 96, kb);
#pragma unroll
      for (int nb = 0; nb < 2; ++nb)
        aV[nb] = wmma_f16(a, vWBv[(kb * 2 + nb) * 32 + lane], aV[nb]);
    }
#pragma unroll
    for (int nb = 0; nb < 2; ++nb) {
#pragma unroll
      for (int g = 0; g < 8; ++g) {
        const int ee = tile * 16 + r0 + g;
        if (ee < E) {
          const int sn = src[ee];
          const int cc = nb * 16 + (lane & 15);
          unsafeAtomicAdd(&outV[((size_t)sn * 32 + cc) * 3 + i],
                          0.1f * aV[nb][g]);
        }
      }
    }
  }

  // ---- psi_d: 9 components, K=80 (pad 96), N=16 ----
#pragma unroll
  for (int ij = 0; ij < 9; ++ij) {
    const int i = ij / 3, j = ij % 3;
    const int i1 = (i + 1) % 3, i2 = (i + 2) % 3;
    wave_sync();
    {
      float y0[8], y1[8], y2[8], y3[8], y4[8];
#pragma unroll
      for (int c8 = 0; c8 < 8; ++c8) {
        const int c = ch0 + c8;
        const float rd = rad[c8];
        const float pd_i0 = pdp[c * 9 + i * 3 + 0];
        const float pd_i1 = pdp[c * 9 + i * 3 + 1];
        const float pd_i2 = pdp[c * 9 + i * 3 + 2];
        const float pdr_i = pd_i0 * rr[0] + pd_i1 * rr[1] + pd_i2 * rr[2];
        const float pd_ij = (j == 0) ? pd_i0 : (j == 1) ? pd_i1 : pd_i2;
        y0[c8] = pa[c8] * rd * rr[i] * rr[j];                         // y022
        y1[c8] = pd_ij * rd;                                          // y202
        y2[c8] = pv[c8][i] * rd * rr[j];                              // y112
        y3[c8] = rd * pdr_i * rr[j];                                  // y222
        y4[c8] = rd * (pdp[c * 9 + i1 * 3 + j] * rr[i2] -
                       pdp[c * 9 + i2 * 3 + j] * rr[i1]);             // y212
      }
      store8h(Yb + m * 96 + 0 + ch0, y0);
      store8h(Yb + m * 96 + 16 + ch0, y1);
      store8h(Yb + m * 96 + 32 + ch0, y2);
      store8h(Yb + m * 96 + 48 + ch0, y3);
      store8h(Yb + m * 96 + 64 + ch0, y4);
    }
    wave_sync();
    v8f aD = zero8;
#pragma unroll
    for (int kb = 0; kb < 3; ++kb) {
      const v16h a = load_A_lds(Yb, 96, kb);
      aD = wmma_f16(a, vWBd[kb * 32 + lane], aD);
    }
#pragma unroll
    for (int g = 0; g < 8; ++g) {
      const int ee = tile * 16 + r0 + g;
      if (ee < E) {
        const int sn = src[ee];
        const int cc = lane & 15;
        unsafeAtomicAdd(&outD[(size_t)sn * 144 + cc * 9 + i * 3 + j],
                        0.1f * aD[g]);
      }
    }
  }
}

// ---------------------------------------------------------------------------
extern "C" void kernel_launch(void* const* d_in, const int* in_sizes, int n_in,
                              void* d_out, int out_size, void* d_ws, size_t ws_size,
                              hipStream_t stream) {
  const float* r_ij = (const float*)d_in[0];
  const float* x_a = (const float*)d_in[1];
  const float* x_v = (const float*)d_in[2];
  const float* x_d = (const float*)d_in[3];
  const float* W_L0 = (const float*)d_in[4];
  const float* W_L1 = (const float*)d_in[5];
  const float* W_L2 = (const float*)d_in[6];
  const float* W_Y000 = (const float*)d_in[7];
  const float* W_Y110 = (const float*)d_in[8];
  const float* W_Y220 = (const float*)d_in[9];
  const float* W_Y011 = (const float*)d_in[10];
  const float* W_Y101 = (const float*)d_in[11];
  const float* W_Y121 = (const float*)d_in[12];
  const float* W_Y211 = (const float*)d_in[13];
  const float* W_Y022 = (const float*)d_in[14];
  const float* W_Y202 = (const float*)d_in[15];
  const float* W_Y112 = (const float*)d_in[16];
  const float* W_Y222 = (const float*)d_in[17];
  const float* W_Y111 = (const float*)d_in[18];
  const float* W_Y212 = (const float*)d_in[19];
  const float* W_enc = (const float*)d_in[20];
  const float* b_enc = (const float*)d_in[21];
  const float* W_dir = (const float*)d_in[22];
  const float* W1 = (const float*)d_in[23];
  const float* b1 = (const float*)d_in[24];
  const float* W2 = (const float*)d_in[25];
  const float* b2 = (const float*)d_in[26];
  const float* W3 = (const float*)d_in[27];
  const float* b3 = (const float*)d_in[28];
  const int* src = (const int*)d_in[29];
  const int* dst = (const int*)d_in[30];
  const int E = in_sizes[29];
  const int N = in_sizes[1] / 64;

  // workspace layout
  float* Pa = (float*)d_ws;
  float* Pv = Pa + (size_t)N * 16;
  float* Pd = Pv + (size_t)N * 48;
  _Float16* WBa = (_Float16*)(Pd + (size_t)N * 144);
  _Float16* WB1 = WBa + 2 * 4 * 512;
  _Float16* WB2 = WB1 + 2 * 8 * 512;
  _Float16* WBdir = WB2 + 4 * 8 * 512;
  _Float16* WB3 = WBdir + 2 * 4 * 512;
  _Float16* WBv = WB3 + 4 * 4 * 512;
  _Float16* WBd = WBv + 3 * 2 * 512;

  float* outA = (float*)d_out;
  float* outV = outA + (size_t)N * 64;
  float* outD = outV + (size_t)N * 96;

  zero_kernel<<<1024, 256, 0, stream>>>((float*)d_out, out_size);

  node_proj_kernel<<<(N * 16 + 127) / 128, 128, 0, stream>>>(
      x_a, x_v, x_d, W_L0, W_L1, W_L2, Pa, Pv, Pd, N);

  // Pack all GEMM weights into WMMA B-operand layout (f16).
  pack_weights<<<32, 256, 0, stream>>>(W_Y000, W_Y110, W_Y220, nullptr, nullptr,
                                       3, 48, 64, 2, 4, WBa);
  pack_weights<<<32, 256, 0, stream>>>(W1, nullptr, nullptr, nullptr, nullptr,
                                       1, 64, 128, 2, 8, WB1);
  pack_weights<<<64, 256, 0, stream>>>(W2, nullptr, nullptr, nullptr, nullptr,
                                       1, 128, 128, 4, 8, WB2);
  pack_weights<<<32, 256, 0, stream>>>(W_dir, nullptr, nullptr, nullptr, nullptr,
                                       1, 64, 64, 2, 4, WBdir);
  pack_weights<<<32, 256, 0, stream>>>(W3, nullptr, nullptr, nullptr, nullptr,
                                       1, 128, 64, 4, 4, WB3);
  pack_weights<<<16, 256, 0, stream>>>(W_Y011, W_Y101, W_Y121, W_Y211, W_Y111,
                                       5, 80, 32, 3, 2, WBv);
  pack_weights<<<8, 256, 0, stream>>>(W_Y022, W_Y202, W_Y112, W_Y222, W_Y212,
                                      5, 80, 16, 3, 1, WBd);

  const int ntiles = (E + 15) / 16;
  const int blocks = (ntiles + WAVES - 1) / WAVES;
  edge_kernel<<<blocks, 128, 0, stream>>>(
      r_ij, dst, src, W_enc, b_enc, b1, b2, b3, Pa, Pv, Pd, WBa, WB1, WB2, WBdir,
      WB3, WBv, WBd, outA, outV, outD, E);
}
